// AttentionHead_65317862637765
// MI455X (gfx1250) — matrix operand order repaired and checked by reading, not compiled, not run on hardware
//
#include <hip/hip_runtime.h>
#include <hip/hip_bf16.h>

// ---------------------------------------------------------------------------
// AttentionHead for MI455X (gfx1250): f16 WMMA flash attention.
//   Stage 1: Q = q@Wq^T+bq, K = k@Wk^T+bk -> f16 row-major in workspace
//            V = v@Wv^T+bv               -> f16 TRANSPOSED (Vt[AD][AN])
//            (all via v_wmma_f32_16x16x32_f16)
//   Stage 2: flash attention over 64-row query blocks; S = Qh Kh^T and
//            O += P V both via WMMA; online softmax 4-way parallel per row.
// ---------------------------------------------------------------------------

typedef __attribute__((ext_vector_type(16))) _Float16 v16h;
typedef __attribute__((ext_vector_type(8)))  _Float16 v8h;
typedef __attribute__((ext_vector_type(8)))  float    v8f;

#define AN 8192
#define AD 256
#define BM 64
#define BN 64
#define SCALE 0.0625f   // 1/sqrt(256)

// ---------------------------------------------------------------------------
// Stage 1: C[m,n] = sum_k X[m,k]*W[n,k] + b[n], output f16.
// One wave per 16x16 tile. TR=false: Out[m][n] row-major [AN][AD].
// TR=true:  Out[n][m] (transposed, [AD][AN]) -- each lane's 8 accumulator
// rows are consecutive m, so the store is one packed 16-byte v8h per lane.
// ---------------------------------------------------------------------------
template <bool TR>
__global__ __launch_bounds__(256) void proj_f16_kernel(
    const float* __restrict__ X, const float* __restrict__ W,
    const float* __restrict__ bias, _Float16* __restrict__ Out)
{
    const int wave = threadIdx.x >> 5;
    const int lane = threadIdx.x & 31;
    const int r15  = lane & 15;
    const int hi   = lane >> 4;

    const int tile = blockIdx.x * (blockDim.x >> 5) + wave;
    const int mt = tile >> 4;      // AD/16 == 16 col tiles
    const int nt = tile & 15;
    const int m0 = mt * 16;
    const int n0 = nt * 16;

    const float* arow = X + (size_t)(m0 + r15) * AD;   // A row m0+r15
    const float* brow = W + (size_t)(n0 + r15) * AD;   // B=W^T col -> W row

    v8f acc = {};
#pragma unroll
    for (int k0 = 0; k0 < AD; k0 += 32) {
        v16h a, b;
#pragma unroll
        for (int i = 0; i < 8; ++i) {
            a[i]     = (_Float16)arow[k0 + hi * 8 + i];        // K = k0+hi*8+i
            a[8 + i] = (_Float16)arow[k0 + 16 + hi * 8 + i];   // K = k0+16+hi*8+i
        }
#pragma unroll
        for (int i = 0; i < 16; ++i)
            b[i] = (_Float16)brow[k0 + (i >> 3) * 16 + hi * 8 + (i & 7)];  // K = k0+(i/8)*16+hi*8+i%8
        acc = __builtin_amdgcn_wmma_f32_16x16x32_f16(
            false, a, false, b, (short)0, acc, false, false);
    }

    const float bv = bias[n0 + r15];
    if (TR) {
        // rows m0+8*hi .. m0+8*hi+7 are contiguous in Vt[n][*]
        v8h pack;
#pragma unroll
        for (int i = 0; i < 8; ++i) pack[i] = (_Float16)(acc[i] + bv);
        *(v8h*)(Out + (size_t)(n0 + r15) * AN + m0 + 8 * hi) = pack;
    } else {
#pragma unroll
        for (int i = 0; i < 8; ++i) {
            const int row = m0 + i + 8 * hi;
            Out[(size_t)row * AD + n0 + r15] = (_Float16)(acc[i] + bv);
        }
    }
}

// ---------------------------------------------------------------------------
// Stage 2: flash attention.
// Block: 256 threads (8 waves), BM=64 query rows, key blocks of BN=64.
// Wave w: row-tile rt = w>>1 (16 rows), half = w&1 (O cols 0..127 / 128..255,
// S col-tiles 2*half..2*half+1).
// Vt is [AD][AN]: B fragments of the PV product are contiguous 32B loads.
// ---------------------------------------------------------------------------
__global__ __launch_bounds__(256) void flash_attn_kernel(
    const _Float16* __restrict__ Qh, const _Float16* __restrict__ Kh,
    const _Float16* __restrict__ Vt, float* __restrict__ Out)
{
    __shared__ __align__(32) float    sS[BM * BN];   // 16 KB scaled scores
    __shared__ __align__(32) _Float16 sP[BM * BN];   //  8 KB probabilities
    __shared__ float sM[BM];                          // running row max
    __shared__ float sL[BM];                          // running row sum
    __shared__ float sCorr[BM];                       // per-iter rescale
    __shared__ float sPmax[BM * 4];                   // partial maxes
    __shared__ float sPsum[BM * 4];                   // partial sums
    __shared__ float sLinv[BM];

    const int tid  = threadIdx.x;
    const int wave = tid >> 5;
    const int lane = tid & 31;
    const int r15  = lane & 15;
    const int hi   = lane >> 4;
    const int rt   = wave >> 1;    // 0..3
    const int half = wave & 1;     // 0..1

    const int m0 = blockIdx.x * BM;

    if (tid < BM) { sM[tid] = -1e30f; sL[tid] = 0.0f; }

    // ---- cache Q fragments for this wave's row tile across the key loop
    const _Float16* qrow = Qh + (size_t)(m0 + rt * 16 + r15) * AD;
    v16h qf[8];
#pragma unroll
    for (int kc = 0; kc < 8; ++kc) {
        const int k0 = kc * 32;
        qf[kc] = *(const v16h*)(qrow + k0 + hi * 16);
    }

    // ---- O accumulators: 8 col tiles of 16 (this wave's 128-col half)
    v8f o[8];
#pragma unroll
    for (int t = 0; t < 8; ++t) o[t] = (v8f){};

    // softmax split: 4 threads per row, 16 columns each
    const int srow  = tid >> 2;
    const int spart = tid & 3;
    const int sbase = srow * BN + spart * 16;

    for (int kb = 0; kb < AN; kb += BN) {
        // prefetch next K / Vt block while computing on this one
        if (kb + BN < AN) {
            __builtin_prefetch((const void*)(Kh + (size_t)(kb + BN + rt * 16) * AD), 0, 1);
            __builtin_prefetch((const void*)(Vt + (size_t)(half * 128 + rt * 16) * AN + kb + BN), 0, 1);
        }

        // ---- Phase 1: S tiles = Qh(rows) @ Kh(cols)^T, scaled, into LDS
#pragma unroll
        for (int cc = 0; cc < 2; ++cc) {
            const int ct = half * 2 + cc;
            const _Float16* krow =
                Kh + (size_t)(kb + ct * 16 + r15) * AD + hi * 16;
            v8f s = {};
#pragma unroll
            for (int kc = 0; kc < 8; ++kc) {
                const v16h b = *(const v16h*)(krow + kc * 32);
                s = __builtin_amdgcn_wmma_f32_16x16x32_f16(
                    false, qf[kc], false, b, (short)0, s, false, false);
            }
#pragma unroll
            for (int i = 0; i < 8; ++i)
                sS[(rt * 16 + i + 8 * hi) * BN + ct * 16 + r15] = s[i] * SCALE;
        }
        __syncthreads();

        // ---- Phase 2a: per-part max (4 threads/row, 16 cols each)
        {
            float pmax = -1e30f;
#pragma unroll
            for (int j = 0; j < 16; ++j) pmax = fmaxf(pmax, sS[sbase + j]);
            sPmax[tid] = pmax;
        }
        __syncthreads();

        // ---- Phase 2b: full row max, exp, partial sum; P -> LDS f16
        float mold, mx, corr;
        {
            mold = sM[srow];
            mx = mold;
#pragma unroll
            for (int p = 0; p < 4; ++p) mx = fmaxf(mx, sPmax[srow * 4 + p]);
            corr = __expf(mold - mx);
            float psum = 0.0f;
#pragma unroll
            for (int j = 0; j < 16; ++j) {
                const float p = __expf(sS[sbase + j] - mx);
                psum += p;
                sP[sbase + j] = (_Float16)p;
            }
            sPsum[tid] = psum;
            if (spart == 0) sCorr[srow] = corr;
        }
        __syncthreads();

        // ---- Phase 2c: part 0 updates running (m, l)
        if (spart == 0) {
            float rs = sPsum[srow * 4] + sPsum[srow * 4 + 1] +
                       sPsum[srow * 4 + 2] + sPsum[srow * 4 + 3];
            sL[srow] = sL[srow] * corr + rs;
            sM[srow] = mx;
        }
        __syncthreads();

        // ---- Phase 3: O = O*corr + P @ V_block
        float c8[8];
#pragma unroll
        for (int i = 0; i < 8; ++i) c8[i] = sCorr[rt * 16 + i + 8 * hi];

        v16h pf[2];
        const _Float16* prow = sP + (rt * 16 + r15) * BN;
#pragma unroll
        for (int kc = 0; kc < 2; ++kc) {
            const int k0 = kc * 32;
            pf[kc] = *(const v16h*)(prow + k0 + hi * 16);
        }

#pragma unroll
        for (int ctile = 0; ctile < 8; ++ctile) {
            const int n0 = half * 128 + ctile * 16;
            v8f acc = o[ctile];
#pragma unroll
            for (int i = 0; i < 8; ++i) acc[i] *= c8[i];
#pragma unroll
            for (int kc = 0; kc < 2; ++kc) {
                // B[kk, n] = Vt[n0+r15][kb + kc*32 + hi*16 + i] -- contiguous!
                const v16h b = *(const v16h*)(
                    Vt + (size_t)(n0 + r15) * AN + kb + kc * 32 + hi * 16);
                acc = __builtin_amdgcn_wmma_f32_16x16x32_f16(
                    false, pf[kc], false, b, (short)0, acc, false, false);
            }
            o[ctile] = acc;
        }
        __syncthreads();
    }

    // ---- epilogue: divide by l and store f32
    if (tid < BM) sLinv[tid] = 1.0f / sL[tid];
    __syncthreads();

    float li[8];
#pragma unroll
    for (int i = 0; i < 8; ++i) li[i] = sLinv[rt * 16 + i + 8 * hi];

#pragma unroll
    for (int ctile = 0; ctile < 8; ++ctile) {
        const int n0 = half * 128 + ctile * 16;
#pragma unroll
        for (int i = 0; i < 8; ++i) {
            const int row = m0 + rt * 16 + i + 8 * hi;
            Out[(size_t)row * AD + n0 + r15] = o[ctile][i] * li[i];
        }
    }
}

// ---------------------------------------------------------------------------
extern "C" void kernel_launch(void* const* d_in, const int* in_sizes, int n_in,
                              void* d_out, int out_size, void* d_ws, size_t ws_size,
                              hipStream_t stream) {
    (void)in_sizes; (void)n_in; (void)out_size; (void)ws_size;
    const float* q  = (const float*)d_in[0];
    const float* k  = (const float*)d_in[1];
    const float* v  = (const float*)d_in[2];
    const float* Wq = (const float*)d_in[3];
    const float* bq = (const float*)d_in[4];
    const float* Wk = (const float*)d_in[5];
    const float* bk = (const float*)d_in[6];
    const float* Wv = (const float*)d_in[7];
    const float* bv = (const float*)d_in[8];
    float* out = (float*)d_out;

    _Float16* Qh = (_Float16*)d_ws;             // 4 MB
    _Float16* Kh = Qh + (size_t)AN * AD;        // 4 MB
    _Float16* Vt = Kh + (size_t)AN * AD;        // 4 MB, [AD][AN] transposed

    // (AN/16)*(AD/16) = 8192 tiles, 8 waves/block -> 1024 blocks
    proj_f16_kernel<false><<<1024, 256, 0, stream>>>(q, Wq, bq, Qh);
    proj_f16_kernel<false><<<1024, 256, 0, stream>>>(k, Wk, bk, Kh);
    proj_f16_kernel<true ><<<1024, 256, 0, stream>>>(v, Wv, bv, Vt);

    flash_attn_kernel<<<AN / BM, 256, 0, stream>>>(Qh, Kh, Vt, out);
}
